// EdgeInfoImgPredictor_20916490732109
// MI455X (gfx1250) — matrix operand order repaired and checked by reading, not compiled, not run on hardware
//
#include <hip/hip_runtime.h>
#include <hip/hip_bf16.h>
#include <math.h>

typedef __attribute__((ext_vector_type(2)))  float    v2f;
typedef __attribute__((ext_vector_type(8)))  float    v8f;
typedef __attribute__((ext_vector_type(16))) _Float16 v16h;

#define DEC_DIM 512
#define LSEL 192
#define BATCH 4
#define ORDER_CLS 17
#define DISPLAY_CLS 15
#define HID_L 1024

// ---------------------------------------------------------------------------
// Stage 1: gathered projections via full-fp32 WMMA (precision-critical, tiny).
//   proj[p][b][t][n],  p = 0:pa_o, 1:pb_o, 2:pa_d, 3:pb_d
//   pa_* projections get the hidden-layer bias b1 folded in (added once).
// One wave computes one 16x16 output tile: M = 16 gathered tokens, N = 16
// output features, K-loop over d=512 in steps of 4 (V_WMMA_F32_16X16X4_F32).
// ---------------------------------------------------------------------------
__global__ __launch_bounds__(256) void proj_kernel(
    const float* __restrict__ hidden,   // (B, 1024, 512)
    const int*   __restrict__ indices,  // (B, 192)
    const float* __restrict__ W1o,      // (1024, 512)
    const float* __restrict__ b1o,      // (512)
    const float* __restrict__ W1d,      // (1024, 512)
    const float* __restrict__ b1d,      // (512)
    float* __restrict__ proj)           // (4, B, 192, 512) workspace
{
    const int wave = (int)((blockIdx.x * blockDim.x + threadIdx.x) >> 5);
    const int lane = (int)(threadIdx.x & 31);
    // wave -> (p:4, mt:48, nt:32); grid sized exactly: 4*48*32 waves
    const int nt = wave & 31;
    const int mt = (wave >> 5) % 48;
    const int p  = wave / (32 * 48);

    const int b  = mt / 12;            // 12 token-tiles of 16 per batch
    const int t0 = (mt % 12) * 16;
    const int n0 = nt * 16;

    const float* W1    = (p < 2) ? W1o : W1d;
    const float* Wpart = W1 + (size_t)(p & 1) * DEC_DIM * DEC_DIM; // rows [0,512) or [512,1024)

    const int m = lane & 15;           // A: row of M-tile / B,D: column
    const int g = lane >> 4;           // K half-select (A/B) and M half-select (D)

    // Gathered A row for this lane (row m of the M-tile)
    const int tok = indices[b * LSEL + t0 + m];
    const float* arow = hidden + ((size_t)b * HID_L + (size_t)tok) * DEC_DIM;

    v8f acc = {};
    for (int k0 = 0; k0 < DEC_DIM; k0 += 4) {
        const int ka = k0 + 2 * g;                 // this lane's K pair
        v2f a;                                     // A[m][ka], A[m][ka+1]
        a.x = arow[ka];
        a.y = arow[ka + 1];
        v2f bm;                                    // B[ka][n0+m], B[ka+1][n0+m]
        bm.x = Wpart[(size_t)ka * DEC_DIM + n0 + m];
        bm.y = Wpart[(size_t)(ka + 1) * DEC_DIM + n0 + m];
        acc = __builtin_amdgcn_wmma_f32_16x16x4_f32(
            false, a, false, bm, (short)0, acc, false, false);
    }

    // Fold b1 into the "pa" (token-i) projections only, per output feature.
    float bias = 0.0f;
    if (p == 0) bias = b1o[n0 + m];
    else if (p == 2) bias = b1d[n0 + m];

    // D layout: VGPR v, lane = n + 16*g  ->  element (row v + 8g, col n)
    float* out = proj + (((size_t)p * BATCH + b) * LSEL + t0) * DEC_DIM;
#pragma unroll
    for (int v = 0; v < 8; ++v) {
        const int r = v + 8 * g;
        out[(size_t)r * DEC_DIM + n0 + m] = acc[v] + bias;
    }
}

// ---------------------------------------------------------------------------
// Stage 2: fused broadcast-add + exact-erf GELU + f16 WMMA contraction.
// One wave owns (head, b, i, j-tile16): M = 16 (i, j) pairs, K = 512 in
// chunks of 32 (V_WMMA_F32_16X16X32_F16, fp32 accumulate). 16-bit WMMA is
// tracked as a TRANS op on CDNA5, so the matrix work co-executes with (and
// hides behind) the erf VALU stream, which is the true bottleneck.
// Branchless masking only — no EXEC manipulation in the hot loop.
// ---------------------------------------------------------------------------
__global__ __launch_bounds__(256) void edge_kernel(
    const float* __restrict__ proj,     // (4, B, 192, 512)
    const float* __restrict__ W2o, const float* __restrict__ b2o,
    const float* __restrict__ W2d, const float* __restrict__ b2d,
    float* __restrict__ out)            // orders (4,17,192,192) ++ displays (4,15,192,192)
{
    const int wave = (int)((blockIdx.x * blockDim.x + threadIdx.x) >> 5);
    const int lane = (int)(threadIdx.x & 31);
    // wave -> (head:2, b:4, i:192, jt:12); grid sized exactly: 18432 waves
    const int jt   = wave % 12;
    const int i    = (wave / 12) % LSEL;
    const int b    = (wave / (12 * LSEL)) % BATCH;
    const int head = wave / (12 * LSEL * BATCH);

    const int j0 = jt * 16;
    const int m  = lane & 15;
    const int g  = lane >> 4;

    // pa row (fixed i, bias pre-folded) and this lane's pb row (j = j0 + m)
    const float* pa = proj + (((size_t)(head * 2 + 0) * BATCH + b) * LSEL + i)      * DEC_DIM;
    const float* pb = proj + (((size_t)(head * 2 + 1) * BATCH + b) * LSEL + j0 + m) * DEC_DIM;

    const float* W2  = head ? W2d : W2o;
    const int    cls = head ? DISPLAY_CLS : ORDER_CLS;
    const int ntile2 = head ? 0 : 1;    // orders needs a second N tile (class 16)

    // Branchless column masks (per-lane constants, no EXEC churn in loop)
    const int   cA  = m;                          // N-tile 0 class
    const int   cB  = 16 + m;                     // N-tile 1 class
    const int   cAi = (cA < cls) ? cA : (cls - 1);
    const int   cBi = (cB < cls) ? cB : (cls - 1);
    const float mA  = (cA < cls) ? 1.0f : 0.0f;
    const float mB  = (cB < cls) ? 1.0f : 0.0f;

    const float kInvSqrt2 = 0.70710678118654752f;

    v8f c0 = {}, c1 = {};
    for (int k0 = 0; k0 < DEC_DIM; k0 += 32) {
        // ---- A operand: 16 exact-erf GELU values in this code's contiguous K order (same as B).
        // lane = m + 16g: elements 0..15 <-> K = k0+16g+s
        v16h a;
#pragma unroll
        for (int half = 0; half < 2; ++half) {
            const int kb = k0 + 16 * g + 8 * half;
#pragma unroll
            for (int t = 0; t < 8; ++t) {
                const float x  = pa[kb + t] + pb[kb + t];
                const float ge = 0.5f * x * (1.0f + erff(x * kInvSqrt2));
                a[half * 8 + t] = (_Float16)ge;
            }
        }

        // ---- B operand(s): W2 block in the same contiguous K order as A.
        // lane = n + 16g: VGPR v holds K = k0+16g+2v (lo), +1 (hi), col n.
        v16h b0;
#pragma unroll
        for (int v = 0; v < 8; ++v) {
            const int kk = k0 + 16 * g + 2 * v;
            b0[2 * v]     = (_Float16)(W2[(size_t)kk       * cls + cAi] * mA);
            b0[2 * v + 1] = (_Float16)(W2[(size_t)(kk + 1) * cls + cAi] * mA);
        }
        c0 = __builtin_amdgcn_wmma_f32_16x16x32_f16(
            false, a, false, b0, (short)0, c0, false, false);

        if (ntile2) {                    // classes 16..31 (only class 16 real)
            v16h b1v;
#pragma unroll
            for (int v = 0; v < 8; ++v) {
                const int kk = k0 + 16 * g + 2 * v;
                b1v[2 * v]     = (_Float16)(W2[(size_t)kk       * cls + cBi] * mB);
                b1v[2 * v + 1] = (_Float16)(W2[(size_t)(kk + 1) * cls + cBi] * mB);
            }
            c1 = __builtin_amdgcn_wmma_f32_16x16x32_f16(
                false, a, false, b1v, (short)0, c1, false, false);
        }
    }

    // Epilogue: add b2, scatter into (b, cls, i, j) layout.
    const float* b2 = head ? b2d : b2o;
    const size_t headbase = head ? (size_t)BATCH * ORDER_CLS * LSEL * LSEL : 0;
#pragma unroll
    for (int v = 0; v < 8; ++v) {
        const int r = v + 8 * g;        // D row -> j offset
        const int j = j0 + r;
        if (cA < cls) {
            const size_t off = headbase + (((size_t)b * cls + cA) * LSEL + i) * LSEL + j;
            out[off] = c0[v] + b2[cA];
        }
        if (ntile2 && (cB < cls)) {
            const size_t off = headbase + (((size_t)b * cls + cB) * LSEL + i) * LSEL + j;
            out[off] = c1[v] + b2[cB];
        }
    }
}

// ---------------------------------------------------------------------------
// Inputs (setup_inputs order):
//  0 img_embed (unused), 1 hidden, 2 indices, 3 W1o, 4 b1o, 5 W2o, 6 b2o,
//  7 W1d, 8 b1d, 9 W2d, 10 b2d
// ---------------------------------------------------------------------------
extern "C" void kernel_launch(void* const* d_in, const int* in_sizes, int n_in,
                              void* d_out, int out_size, void* d_ws, size_t ws_size,
                              hipStream_t stream) {
    const float* hidden  = (const float*)d_in[1];
    const int*   indices = (const int*)  d_in[2];
    const float* W1o     = (const float*)d_in[3];
    const float* b1o     = (const float*)d_in[4];
    const float* W2o     = (const float*)d_in[5];
    const float* b2o     = (const float*)d_in[6];
    const float* W1d     = (const float*)d_in[7];
    const float* b1d     = (const float*)d_in[8];
    const float* W2d     = (const float*)d_in[9];
    const float* b2d     = (const float*)d_in[10];

    float* proj = (float*)d_ws;  // 4 * 4 * 192 * 512 floats = 6 MB scratch

    // Stage 1: 4 proj * 48 m-tiles * 32 n-tiles = 6144 waves -> 768 blocks
    proj_kernel<<<768, 256, 0, stream>>>(hidden, indices, W1o, b1o, W1d, b1d, proj);

    // Stage 2: 2 heads * 4 b * 192 i * 12 j-tiles = 18432 waves -> 2304 blocks
    edge_kernel<<<2304, 256, 0, stream>>>(proj, W2o, b2o, W2d, b2d, (float*)d_out);
}
